// MHA_103079215226
// MI455X (gfx1250) — matrix operand order repaired and checked by reading, not compiled, not run on hardware
//
#include <hip/hip_runtime.h>
#include <hip/hip_bf16.h>

typedef __bf16 bf16_t;
typedef __attribute__((ext_vector_type(16))) __bf16 v16bf;
typedef __attribute__((ext_vector_type(8)))  __bf16 v8bf;
typedef __attribute__((ext_vector_type(4)))  __bf16 v4bf;
typedef __attribute__((ext_vector_type(8)))  float  v8f;

#define DMODEL 1024
#define NHEAD  16
#define DHEAD  64
#define SEQ    2048
#define NB     2
#define MROWS  (NB * SEQ)        // 4096

union FragBF { v16bf v; v8bf h2[2]; };

// ---------------------------------------------------------------------------
// fp32 -> bf16 elementwise convert (vectorized x4)
// ---------------------------------------------------------------------------
__global__ void cvt_f32_bf16(const float* __restrict__ src, bf16_t* __restrict__ dst, int n4) {
  int i = blockIdx.x * blockDim.x + threadIdx.x;
  int stride = gridDim.x * blockDim.x;
  for (; i < n4; i += stride) {
    float4 f = ((const float4*)src)[i];
    v4bf o = { (bf16_t)f.x, (bf16_t)f.y, (bf16_t)f.z, (bf16_t)f.w };
    ((v4bf*)dst)[i] = o;
  }
}

// ---------------------------------------------------------------------------
// Weight transpose+convert: W[1024][1024] f32 row-major -> WT[1024][1024] bf16
// (WT[n][k] = W[k][n]).  Tiled through LDS for coalescing.
// ---------------------------------------------------------------------------
__global__ __launch_bounds__(256)
void cvt_transpose_f32_bf16(const float* __restrict__ src, bf16_t* __restrict__ dst) {
  __shared__ float tile[32][33];
  int bx = blockIdx.x * 32;      // src col block
  int by = blockIdx.y * 32;      // src row block
  int tx = threadIdx.x & 31;
  int ty = threadIdx.x >> 5;     // 0..7
  #pragma unroll
  for (int i = ty; i < 32; i += 8)
    tile[i][tx] = src[(size_t)(by + i) * DMODEL + bx + tx];
  __syncthreads();
  #pragma unroll
  for (int i = ty; i < 32; i += 8)
    dst[(size_t)(bx + i) * DMODEL + by + tx] = (bf16_t)tile[tx][i];
}

// ---------------------------------------------------------------------------
// bf16 WMMA GEMM:  C[4096,1024] = A[4096,1024] @ W + bias, with W given
// pre-transposed (WT[n][k]).  Block tile 64x128, 8 waves, 32x32 per wave.
// A tile staged via CDNA5 async DMA (global -> LDS, ASYNCcnt); B fragments are
// contiguous 32B per-lane global loads out of the L2-resident transposed W.
// OUT_MODE: 0 = f32 row-major, 1 = bf16 row-major, 2 = bf16 transposed [N][4096]
// ---------------------------------------------------------------------------
template<int OUT_MODE>
__global__ __launch_bounds__(256)
void gemm_bf16_1024(const bf16_t* __restrict__ A, const bf16_t* __restrict__ BT,
                    const float* __restrict__ bias,
                    bf16_t* __restrict__ Cbf, float* __restrict__ Cf) {
  const int K = 1024, N = 1024;
  __shared__ __attribute__((aligned(32))) bf16_t As[64][32];

  int tid  = threadIdx.x;
  int lane = tid & 31;
  int w    = tid >> 5;       // 0..7
  int wm   = w >> 2;         // 0..1
  int wn   = w & 3;          // 0..3
  int half = lane >> 4;      // 0 or 1
  int l16  = lane & 15;
  int rowblk = blockIdx.x * 64;
  int colblk = blockIdx.y * 128;

  // A staging coordinates (fixed for whole K loop): 256 lanes x 16B = 64x32 tile
  int ar = tid >> 2;
  int ac = (tid & 3) * 8;
  unsigned a_lds = (unsigned)(uintptr_t)&As[ar][ac];   // LDS offset in low 32 bits
  const bf16_t* a_gp = &A[(size_t)(rowblk + ar) * K + ac];

  // B fragment base pointers: lane reads WT[col][k0+bkb .. +15] (32B contiguous)
  int bkb = half ? 16 : 0;
  const bf16_t* bt0 = &BT[(size_t)(colblk + wn * 32 + l16) * K + bkb];
  const bf16_t* bt1 = bt0 + (size_t)16 * K;

  v8f acc[2][2] = {};

  for (int k0 = 0; k0 < K; k0 += 32) {
    // async DMA: global -> LDS, no VGPR roundtrip
    asm volatile("global_load_async_to_lds_b128 %0, %1, off"
                 :: "v"(a_lds), "v"(a_gp + k0) : "memory");
    v16bf bfr0 = *(const v16bf*)(bt0 + k0);
    v16bf bfr1 = *(const v16bf*)(bt1 + k0);
    asm volatile("s_wait_asynccnt 0x0" ::: "memory");
    __syncthreads();

    FragBF afr[2];
    int akb = half ? 16 : 0;
    #pragma unroll
    for (int i = 0; i < 2; ++i) {
      int r = wm * 32 + i * 16 + l16;
      afr[i].h2[0] = *(const v8bf*)&As[r][akb];
      afr[i].h2[1] = *(const v8bf*)&As[r][akb + 8];
    }
    acc[0][0] = __builtin_amdgcn_wmma_f32_16x16x32_bf16(false, afr[0].v, false, bfr0, (short)0, acc[0][0], false, false);
    acc[0][1] = __builtin_amdgcn_wmma_f32_16x16x32_bf16(false, afr[0].v, false, bfr1, (short)0, acc[0][1], false, false);
    acc[1][0] = __builtin_amdgcn_wmma_f32_16x16x32_bf16(false, afr[1].v, false, bfr0, (short)0, acc[1][0], false, false);
    acc[1][1] = __builtin_amdgcn_wmma_f32_16x16x32_bf16(false, afr[1].v, false, bfr1, (short)0, acc[1][1], false, false);
    __syncthreads();
  }

  #pragma unroll
  for (int i = 0; i < 2; ++i) {
    #pragma unroll
    for (int j = 0; j < 2; ++j) {
      int col = colblk + wn * 32 + j * 16 + l16;
      float bv = bias[col];
      #pragma unroll
      for (int r8 = 0; r8 < 8; ++r8) {
        int row = rowblk + wm * 32 + i * 16 + half * 8 + r8;
        float vout = acc[i][j][r8] + bv;
        if constexpr (OUT_MODE == 0)      Cf [(size_t)row * N + col] = vout;
        else if constexpr (OUT_MODE == 1) Cbf[(size_t)row * N + col] = (bf16_t)vout;
        else                              Cbf[(size_t)col * MROWS + row] = (bf16_t)vout;
      }
    }
  }
}

// ---------------------------------------------------------------------------
// Attention scores + two-pass softmax.
// One block = one (n, h, 16-query tile). 8 waves split the 128 key tiles.
// Pass 1: per-lane online (max,sum) -- NO cross-lane traffic in the tile loop;
// the 16-lane log-sum-exp merge happens once, then the 8-wave merge in LDS.
// Pass 2: recompute S, write normalized fp32 P to d_out exactly once.
// ---------------------------------------------------------------------------
__global__ __launch_bounds__(256)
void attn_scores_softmax(const bf16_t* __restrict__ Qp, const bf16_t* __restrict__ Kp,
                         float* __restrict__ P) {
  int b  = blockIdx.x;
  int qt = b & 127;
  int h  = (b >> 7) & 15;
  int n  = b >> 11;
  int tid = threadIdx.x, lane = tid & 31, w = tid >> 5;
  int half = lane >> 4, l16 = lane & 15;
  const float inv_temp = 0.125f;   // 1/sqrt(64)

  __shared__ float red_m[8][16];
  __shared__ float red_s[8][16];
  __shared__ float fin_m[16];
  __shared__ float fin_is[16];

  // Q fragments (A-matrix 16x32, two dh halves), held in registers
  FragBF qa[2];
  {
    const bf16_t* qrow = Qp + ((size_t)(n * SEQ + qt * 16 + l16)) * DMODEL + h * DHEAD;
    int kb = half ? 16 : 0;
    #pragma unroll
    for (int f = 0; f < 2; ++f) {
      qa[f].h2[0] = *(const v8bf*)&qrow[f * 32 + kb];
      qa[f].h2[1] = *(const v8bf*)&qrow[f * 32 + kb + 8];
    }
  }

  const bf16_t* kbase = Kp + ((size_t)n * SEQ) * DMODEL + h * DHEAD + (half ? 16 : 0);

  float m[8], ssum[8];
  #pragma unroll
  for (int j = 0; j < 8; ++j) { m[j] = -1e30f; ssum[j] = 0.f; }

  // ---- pass 1: per-lane online max/sum (no shuffles inside the loop) ----
  for (int kt = w; kt < 128; kt += 8) {
    const bf16_t* krow = kbase + (size_t)(kt * 16 + l16) * DMODEL;
    v16bf kb0 = *(const v16bf*)&krow[0];
    v16bf kb1 = *(const v16bf*)&krow[32];
    v8f s = {};
    s = __builtin_amdgcn_wmma_f32_16x16x32_bf16(false, qa[0].v, false, kb0, (short)0, s, false, false);
    s = __builtin_amdgcn_wmma_f32_16x16x32_bf16(false, qa[1].v, false, kb1, (short)0, s, false, false);
    #pragma unroll
    for (int j = 0; j < 8; ++j) {
      float v = s[j] * inv_temp;
      float mnew = fmaxf(m[j], v);
      ssum[j] = ssum[j] * __expf(m[j] - mnew) + __expf(v - mnew);
      m[j] = mnew;
    }
  }

  // ---- 16-lane (m,s) log-sum-exp merge, once ----
  #pragma unroll
  for (int off = 1; off < 16; off <<= 1) {
    #pragma unroll
    for (int j = 0; j < 8; ++j) {
      float mo = __shfl_xor(m[j], off, 32);
      float so = __shfl_xor(ssum[j], off, 32);
      float mn = fmaxf(m[j], mo);
      ssum[j] = ssum[j] * __expf(m[j] - mn) + so * __expf(mo - mn);
      m[j] = mn;
    }
  }

  // ---- merge 8 waves' (m, s) via LDS ----
  if (l16 == 0) {
    #pragma unroll
    for (int j = 0; j < 8; ++j) {
      red_m[w][half * 8 + j] = m[j];
      red_s[w][half * 8 + j] = ssum[j];
    }
  }
  __syncthreads();
  if (tid < 16) {
    float M = -1e30f;
    for (int ww = 0; ww < 8; ++ww) M = fmaxf(M, red_m[ww][tid]);
    float S = 0.f;
    for (int ww = 0; ww < 8; ++ww) S += red_s[ww][tid] * __expf(red_m[ww][tid] - M);
    fin_m[tid]  = M;
    fin_is[tid] = 1.0f / S;
  }
  __syncthreads();

  float Mrow[8], ISrow[8];
  #pragma unroll
  for (int j = 0; j < 8; ++j) {
    Mrow[j]  = fin_m[half * 8 + j];
    ISrow[j] = fin_is[half * 8 + j];
  }

  // ---- pass 2: recompute S, write normalized P ----
  float* Pbase = P + (((size_t)(n * NHEAD + h)) * SEQ + qt * 16) * SEQ;
  for (int kt = w; kt < 128; kt += 8) {
    const bf16_t* krow = kbase + (size_t)(kt * 16 + l16) * DMODEL;
    v16bf kb0 = *(const v16bf*)&krow[0];
    v16bf kb1 = *(const v16bf*)&krow[32];
    v8f s = {};
    s = __builtin_amdgcn_wmma_f32_16x16x32_bf16(false, qa[0].v, false, kb0, (short)0, s, false, false);
    s = __builtin_amdgcn_wmma_f32_16x16x32_bf16(false, qa[1].v, false, kb1, (short)0, s, false, false);
    #pragma unroll
    for (int j = 0; j < 8; ++j) {
      float v = s[j] * inv_temp;
      float p = __expf(v - Mrow[j]) * ISrow[j];
      Pbase[(size_t)(half * 8 + j) * SEQ + kt * 16 + l16] = p;
    }
  }
}

// ---------------------------------------------------------------------------
// agg = P @ Vproj, with Vproj stored TRANSPOSED (Vt[dmodel_col][4096 rows]).
// One block = (n, h, 128-query slab); 8 waves x 16 queries.
// No LDS, no barriers: both A (P rows) and B (Vt columns) fragments are
// contiguous per-lane loads.
// ---------------------------------------------------------------------------
__global__ __launch_bounds__(256)
void attn_agg(const float* __restrict__ P, const bf16_t* __restrict__ Vt,
              float* __restrict__ agg) {
  int b = blockIdx.x;
  int qt128 = b & 15;
  int h = (b >> 4) & 15;
  int n = b >> 8;
  int tid = threadIdx.x, lane = tid & 31, w = tid >> 5;
  int half = lane >> 4, l16 = lane & 15;

  int qbase = qt128 * 128 + w * 16;
  const float* Prow = P + (((size_t)(n * NHEAD + h)) * SEQ + qbase + l16) * SEQ;
  int bkb = half ? 16 : 0;
  const bf16_t* vcol = Vt + ((size_t)(h * DHEAD + l16)) * MROWS + n * SEQ + bkb;

  v8f acc[4] = {};

  for (int k0 = 0; k0 < SEQ; k0 += 32) {
    // A fragment from fp32 P, converted to bf16
    FragBF af;
    {
      int kb = half ? 16 : 0;
      v8f p0 = *(const v8f*)&Prow[k0 + kb];
      v8f p1 = *(const v8f*)&Prow[k0 + kb + 8];
      #pragma unroll
      for (int i = 0; i < 8; ++i) {
        af.h2[0][i] = (bf16_t)p0[i];
        af.h2[1][i] = (bf16_t)p1[i];
      }
    }
    #pragma unroll
    for (int j = 0; j < 4; ++j) {
      v16bf bf = *(const v16bf*)&vcol[(size_t)(j * 16) * MROWS + k0];
      acc[j] = __builtin_amdgcn_wmma_f32_16x16x32_bf16(
          false, af.v, false, bf, (short)0, acc[j], false, false);
    }
  }

  #pragma unroll
  for (int j = 0; j < 4; ++j)
    #pragma unroll
    for (int r = 0; r < 8; ++r) {
      int g = qbase + half * 8 + r;
      agg[((size_t)(n * SEQ + g)) * DMODEL + h * DHEAD + j * 16 + l16] = acc[j][r];
    }
}

// ---------------------------------------------------------------------------
// Row-wise L2 normalize (tf.nn.l2_normalize semantics), write bf16 for GEMM.
// ---------------------------------------------------------------------------
__global__ __launch_bounds__(256)
void l2norm_rows(const float* __restrict__ x, bf16_t* __restrict__ y) {
  int row = blockIdx.x;
  const float* xr = x + (size_t)row * DMODEL;
  float s = 0.f;
  for (int i = threadIdx.x; i < DMODEL; i += 256) { float v = xr[i]; s += v * v; }
  __shared__ float red[256];
  red[threadIdx.x] = s;
  __syncthreads();
  for (int off = 128; off > 0; off >>= 1) {
    if (threadIdx.x < off) red[threadIdx.x] += red[threadIdx.x + off];
    __syncthreads();
  }
  float scale = rsqrtf(fmaxf(red[0], 1e-12f));
  bf16_t* yr = y + (size_t)row * DMODEL;
  for (int i = threadIdx.x; i < DMODEL; i += 256) yr[i] = (bf16_t)(xr[i] * scale);
}

// ---------------------------------------------------------------------------
extern "C" void kernel_launch(void* const* d_in, const int* in_sizes, int n_in,
                              void* d_out, int out_size, void* d_ws, size_t ws_size,
                              hipStream_t stream) {
  (void)in_sizes; (void)n_in; (void)out_size; (void)ws_size;
  const float* q  = (const float*)d_in[0];
  const float* k  = (const float*)d_in[1];
  const float* v  = (const float*)d_in[2];
  const float* Wk = (const float*)d_in[3];
  const float* bk = (const float*)d_in[4];
  const float* Wv = (const float*)d_in[5];
  const float* bv = (const float*)d_in[6];
  const float* Wo = (const float*)d_in[7];
  const float* bo = (const float*)d_in[8];

  float* out = (float*)d_out;
  float* P   = out + (size_t)NB * SEQ * DMODEL;   // soft_attention region

  char* ws = (char*)d_ws;
  const size_t MB = 1024 * 1024;
  bf16_t* qb  = (bf16_t*)(ws);
  bf16_t* kb  = (bf16_t*)(ws +  8 * MB);
  bf16_t* vb  = (bf16_t*)(ws + 16 * MB);
  bf16_t* WkT = (bf16_t*)(ws + 24 * MB);
  bf16_t* WvT = (bf16_t*)(ws + 26 * MB);
  bf16_t* WoT = (bf16_t*)(ws + 28 * MB);
  bf16_t* Qp  = (bf16_t*)(ws + 30 * MB);
  bf16_t* Kp  = (bf16_t*)(ws + 38 * MB);
  bf16_t* Vt  = (bf16_t*)(ws + 46 * MB);   // transposed V projection [DMODEL][4096]
  float*  agg = (float*) (ws + 54 * MB);
  bf16_t* nrm = (bf16_t*)(ws + 70 * MB);

  const int nAct4 = NB * SEQ * DMODEL / 4;      // 1,048,576

  cvt_f32_bf16<<<1024, 256, 0, stream>>>(q, qb, nAct4);
  cvt_f32_bf16<<<1024, 256, 0, stream>>>(k, kb, nAct4);
  cvt_f32_bf16<<<1024, 256, 0, stream>>>(v, vb, nAct4);

  dim3 tg(32, 32);
  cvt_transpose_f32_bf16<<<tg, 256, 0, stream>>>(Wk, WkT);
  cvt_transpose_f32_bf16<<<tg, 256, 0, stream>>>(Wv, WvT);
  cvt_transpose_f32_bf16<<<tg, 256, 0, stream>>>(Wo, WoT);

  dim3 gg(64, 8);   // M=4096 / 64, N=1024 / 128
  gemm_bf16_1024<1><<<gg, 256, 0, stream>>>(qb, WkT, bk, Qp, nullptr);  // q uses fc_k
  gemm_bf16_1024<1><<<gg, 256, 0, stream>>>(kb, WkT, bk, Kp, nullptr);
  gemm_bf16_1024<2><<<gg, 256, 0, stream>>>(vb, WvT, bv, Vt, nullptr);  // transposed out

  attn_scores_softmax<<<NB * NHEAD * (SEQ / 16), 256, 0, stream>>>(Qp, Kp, P);
  attn_agg<<<NB * NHEAD * (SEQ / 128), 256, 0, stream>>>(P, Vt, agg);
  l2norm_rows<<<NB * SEQ, 256, 0, stream>>>(agg, nrm);

  gemm_bf16_1024<0><<<gg, 256, 0, stream>>>(nrm, WoT, bo, nullptr, out);
}
